// GRUNetwork_83983790506229
// MI455X (gfx1250) — matrix operand order repaired and checked by reading, not compiled, not run on hardware
//
#include <hip/hip_runtime.h>
#include <hip/hip_bf16.h>
#include <math.h>

// ---------------------------------------------------------------------------
// GRU network for MI455X (gfx1250, wave32, WMMA).
// Layer 0 dominates: sequential GRU, per-step 6144x2048 GEMV vs L2-resident
// bf16 weights (25 MB/step). Input projections are hoisted into one big
// WMMA GEMM. Layers 1/2 degenerate to single matvecs (h=0 => Whh unused).
// ---------------------------------------------------------------------------

typedef __attribute__((ext_vector_type(16))) __bf16 v16bf;
typedef __attribute__((ext_vector_type(8)))  float  v8f;
typedef __attribute__((ext_vector_type(4)))  unsigned int v4u;

#define HIDDEN 2048
#define H3     6144
#define T_SEQ  4096
#define IN_SZ  32
#define OUT_SZ 3

// ---- workspace layout (bytes) ---------------------------------------------
#define WS_PACKW   0u                        // 6144*2048 bf16 = 25165824
#define WS_XP      25165824u                 // 4096*6144 bf16 = 50331648
#define WS_HB      75497472u                 // 2*2048 bf16    = 8192
#define WS_HF      75505664u                 // 2*2048 f32     = 16384
#define WS_A1      75522048u                 // 6144 f32       = 24576
#define WS_H1      75546624u                 // 2048 f32       = 8192
#define WS_A2      75554816u                 // 6144 f32       = 24576
#define WS_H2      75579392u                 // 2048 f32       = 8192
#define WS_BAR     75587584u                 // barrier counter

// A-operand (16-bit, 16x32) K index for (vgpr v, lane-half): per CDNA5 ISA 7.12.2
__device__ __forceinline__ int a_k0(int v, int half) {
    return 2 * v + (v >= 4 ? 8 : 0) + 8 * half;
}
// B-operand as loaded here (16-bit, 32x16): lane half h holds K = 16*h + (0..15), matching packed A.

// ---------------------------------------------------------------------------
// init: zero barrier + h double buffers (ws is poisoned 0xAA by harness)
__global__ void init_ws_kernel(unsigned* bar, __bf16* hb, float* hf) {
    if (threadIdx.x == 0) *bar = 0u;
    for (int i = threadIdx.x; i < 2 * HIDDEN; i += blockDim.x) {
        hb[i] = (__bf16)0.0f;
        hf[i] = 0.0f;
    }
}

// ---------------------------------------------------------------------------
// Repack Whh0 (fp32 row-major [6144][2048]) into bf16 WMMA-A-swizzled tiles.
// Tile index = ((g*3 + s)*64 + kk), g=row-group(128), s=gate section(3),
// kk=k-chunk(64). Each (tile, lane) owns 16 contiguous bf16 (32 B).
__global__ void pack_whh_kernel(const float* __restrict__ W, __bf16* __restrict__ P) {
    int idx  = blockIdx.x * blockDim.x + threadIdx.x;   // 128*3*64*32 threads
    int lane = idx & 31;
    int tile = idx >> 5;
    int kk = tile & 63;
    int gs = tile >> 6;
    int s = gs % 3, g = gs / 3;
    int M = lane & 15, half = lane >> 4;
    int row = s * HIDDEN + g * 16 + M;
    const float* wr = W + (size_t)row * HIDDEN + kk * 32;
    __bf16* out = P + ((size_t)tile * 32 + lane) * 16;
#pragma unroll
    for (int v = 0; v < 8; ++v) {
        int kc = 2 * v + 16 * half;
        out[2 * v]     = (__bf16)wr[kc];
        out[2 * v + 1] = (__bf16)wr[kc + 1];
    }
}

// ---------------------------------------------------------------------------
// Xp[t, j] = Wih0 @ x_t + b0, as WMMA GEMM. K=32 => one wmma per 16x16 tile.
// One wave per tile: 256 t-tiles x 384 j-tiles.
__global__ void xproj_kernel(const float* __restrict__ x, const float* __restrict__ Wih0,
                             const float* __restrict__ b0, __bf16* __restrict__ Xp) {
    int wid  = (blockIdx.x * blockDim.x + threadIdx.x) >> 5;
    int lane = threadIdx.x & 31;
    int ji = wid % 384, ti = wid / 384;
    int M = lane & 15, half = lane >> 4, N = M;

    v16bf a, b;
    const float* xr = x + (size_t)(ti * 16 + M) * IN_SZ;
#pragma unroll
    for (int v = 0; v < 8; ++v) {
        int k = 2 * v + 16 * half;
        a[2 * v]     = (__bf16)xr[k];
        a[2 * v + 1] = (__bf16)xr[k + 1];
    }
    const float* wc = Wih0 + (size_t)(ji * 16 + N) * IN_SZ;  // B(k,n) = Wih0[n, k]
#pragma unroll
    for (int v = 0; v < 8; ++v) {
        int k = 2 * v + 16 * half;
        b[2 * v]     = (__bf16)wc[k];
        b[2 * v + 1] = (__bf16)wc[k + 1];
    }
    v8f c = {};
    c = __builtin_amdgcn_wmma_f32_16x16x32_bf16(false, a, false, b, (short)0, c, false, false);
    float bias = b0[ji * 16 + N];
#pragma unroll
    for (int r = 0; r < 8; ++r) {
        int trow = ti * 16 + r + 8 * half;
        Xp[(size_t)trow * H3 + ji * 16 + N] = (__bf16)(c[r] + bias);
    }
}

// ---------------------------------------------------------------------------
// Persistent sequential GRU. 128 WGs x 256 threads (8 waves). WG g owns rows
// j in [16g, 16g+16) for all three gates; waves split K 8-ways; one grid
// barrier per step (monotonic atomic counter). h double-buffered in global.
__global__ void __launch_bounds__(256) gru_seq_kernel(
        const __bf16* __restrict__ Pw, const __bf16* __restrict__ Xp,
        const float* __restrict__ bn0, __bf16* hb, float* hf,
        unsigned* bar, int T) {
    __shared__ __bf16 hsh[HIDDEN];          // staged h_t (4 KB)
    __shared__ float  part[8][3][16];       // per-wave partial gate rows

    const int tid  = threadIdx.x;
    const int lane = tid & 31;
    const int w    = tid >> 5;
    const int g    = blockIdx.x;            // 0..127
    const int jb   = g * 16;
    const int M    = lane & 15, half = lane >> 4;
    const unsigned nwg = gridDim.x;

    for (int t = 0; t < T; ++t) {
        const __bf16* hb_old = hb + (t & 1) * HIDDEN;
        // stage h -> LDS (16 B per thread)
        *(v4u*)&hsh[tid * 8] = *(const v4u*)&hb_old[tid * 8];
        __syncthreads();

        v8f cr = {}, cz = {}, cn = {};
#pragma unroll
        for (int i = 0; i < 8; ++i) {
            int kk = w * 8 + i;
            // B operand: 32 contiguous bytes of h
            const v4u* hp = (const v4u*)&hsh[kk * 32 + 16 * half];
            v16bf bm;
            ((v4u*)&bm)[0] = hp[0];
            ((v4u*)&bm)[1] = hp[1];
            // A operands: pre-swizzled bf16 weight tiles (32 B per lane)
            const v4u* Ar = (const v4u*)(Pw + (((size_t)(g * 3 + 0) * 64 + kk) * 32 + lane) * 16);
            const v4u* Az = (const v4u*)(Pw + (((size_t)(g * 3 + 1) * 64 + kk) * 32 + lane) * 16);
            const v4u* An = (const v4u*)(Pw + (((size_t)(g * 3 + 2) * 64 + kk) * 32 + lane) * 16);
            v16bf ar, az, an;
            ((v4u*)&ar)[0] = Ar[0]; ((v4u*)&ar)[1] = Ar[1];
            ((v4u*)&az)[0] = Az[0]; ((v4u*)&az)[1] = Az[1];
            ((v4u*)&an)[0] = An[0]; ((v4u*)&an)[1] = An[1];
            if (i < 7)  // prefetch next r-tile line -> global_prefetch
                __builtin_prefetch((const void*)(Ar + 64), 0, 1);
            cr = __builtin_amdgcn_wmma_f32_16x16x32_bf16(false, ar, false, bm, (short)0, cr, false, false);
            cz = __builtin_amdgcn_wmma_f32_16x16x32_bf16(false, az, false, bm, (short)0, cz, false, false);
            cn = __builtin_amdgcn_wmma_f32_16x16x32_bf16(false, an, false, bm, (short)0, cn, false, false);
        }
        // B columns replicated => lane 0 holds rows 0..7, lane 16 rows 8..15
        if (M == 0) {
#pragma unroll
            for (int r = 0; r < 8; ++r) {
                part[w][0][r + 8 * half] = cr[r];
                part[w][1][r + 8 * half] = cz[r];
                part[w][2][r + 8 * half] = cn[r];
            }
        }
        __syncthreads();

        if (tid < 16) {  // gate math for this WG's 16 rows
            int j = jb + tid;
            float ir = 0.f, iz = 0.f, in = 0.f;
#pragma unroll
            for (int ww = 0; ww < 8; ++ww) {
                ir += part[ww][0][tid];
                iz += part[ww][1][tid];
                in += part[ww][2][tid];
            }
            const __bf16* xp = Xp + (size_t)t * H3;
            float xr = (float)xp[j];
            float xz = (float)xp[HIDDEN + j];
            float xn = (float)xp[2 * HIDDEN + j];
            float hold = hf[(t & 1) * HIDDEN + j];
            float r = 1.0f / (1.0f + __expf(-(xr + ir)));
            float z = 1.0f / (1.0f + __expf(-(xz + iz)));
            float n = tanhf(xn + r * (in + bn0[j]));
            float hnew = n + z * (hold - n);
            hf[((t + 1) & 1) * HIDDEN + j] = hnew;
            hb[((t + 1) & 1) * HIDDEN + j] = (__bf16)hnew;
        }
        // --- grid barrier (release h, acquire before next staging) ---
        __threadfence();
        __syncthreads();
        if (tid == 0) {
            unsigned target = (unsigned)(t + 1) * nwg;
            atomicAdd(bar, 1u);
            while (atomicAdd(bar, 0u) < target) __builtin_amdgcn_s_sleep(1);
        }
        __syncthreads();
        __threadfence();
    }
}

// ---------------------------------------------------------------------------
// one wave per row: out[row] = W[row,:] . (relu?)vin
__global__ void matvec_rows_kernel(const float* __restrict__ W, const float* __restrict__ vin,
                                   float* __restrict__ out, int rows, int K, int relu_in) {
    int wid  = (blockIdx.x * blockDim.x + threadIdx.x) >> 5;
    int lane = threadIdx.x & 31;
    if (wid >= rows) return;
    const float* wr = W + (size_t)wid * K;
    float s = 0.f;
    for (int k = lane; k < K; k += 32) {
        float v = vin[k];
        if (relu_in) v = v > 0.f ? v : 0.f;
        s += wr[k] * v;
    }
#pragma unroll
    for (int off = 16; off; off >>= 1) s += __shfl_down(s, off, 32);
    if (lane == 0) out[wid] = s;
}

// single GRU step from h=0:  h = n*(1-z)
__global__ void gru_gate0_kernel(const float* __restrict__ a, const float* __restrict__ b,
                                 const float* __restrict__ bn, float* __restrict__ hout) {
    int j = blockIdx.x * blockDim.x + threadIdx.x;
    if (j >= HIDDEN) return;
    float ir = a[j] + b[j];
    float iz = a[HIDDEN + j] + b[HIDDEN + j];
    float inn = a[2 * HIDDEN + j] + b[2 * HIDDEN + j];
    float r = 1.0f / (1.0f + __expf(-ir));
    float z = 1.0f / (1.0f + __expf(-iz));
    float n = tanhf(inn + r * bn[j]);
    hout[j] = n * (1.0f - z);
}

__global__ void out_mv_kernel(const float* __restrict__ Wout, const float* __restrict__ h,
                              const float* __restrict__ bout, float* __restrict__ out) {
    int wid  = threadIdx.x >> 5;
    int lane = threadIdx.x & 31;
    if (wid >= OUT_SZ) return;
    const float* wr = Wout + (size_t)wid * HIDDEN;
    float s = 0.f;
    for (int k = lane; k < HIDDEN; k += 32) s += wr[k] * h[k];
#pragma unroll
    for (int off = 16; off; off >>= 1) s += __shfl_down(s, off, 32);
    if (lane == 0) out[wid] = bout[wid] + s;
}

// ---------------------------------------------------------------------------
extern "C" void kernel_launch(void* const* d_in, const int* in_sizes, int n_in,
                              void* d_out, int out_size, void* d_ws, size_t ws_size,
                              hipStream_t stream) {
    const float* x    = (const float*)d_in[0];
    const float* Wih0 = (const float*)d_in[1];
    const float* Whh0 = (const float*)d_in[2];
    const float* b0   = (const float*)d_in[3];
    const float* bn0  = (const float*)d_in[4];
    const float* Wih1 = (const float*)d_in[5];
    // d_in[6] = Whh1 (unused: h=0 for the single step)
    const float* b1   = (const float*)d_in[7];
    const float* bn1  = (const float*)d_in[8];
    const float* Wih2 = (const float*)d_in[9];
    // d_in[10] = Whh2 (unused)
    const float* b2   = (const float*)d_in[11];
    const float* bn2  = (const float*)d_in[12];
    const float* Wout = (const float*)d_in[13];
    const float* bout = (const float*)d_in[14];
    float* out = (float*)d_out;

    char* ws = (char*)d_ws;
    __bf16*   packW = (__bf16*)(ws + WS_PACKW);
    __bf16*   Xp    = (__bf16*)(ws + WS_XP);
    __bf16*   hb    = (__bf16*)(ws + WS_HB);
    float*    hf    = (float*)(ws + WS_HF);
    float*    a1    = (float*)(ws + WS_A1);
    float*    h1    = (float*)(ws + WS_H1);
    float*    a2    = (float*)(ws + WS_A2);
    float*    h2    = (float*)(ws + WS_H2);
    unsigned* bar   = (unsigned*)(ws + WS_BAR);

    init_ws_kernel<<<1, 256, 0, stream>>>(bar, hb, hf);
    // repack Whh0: 128*3*64 tiles * 32 lanes = 786432 threads
    pack_whh_kernel<<<786432 / 256, 256, 0, stream>>>(Whh0, packW);
    // input projection GEMM: 256*384 tiles, one wave each => 12288 blocks
    xproj_kernel<<<(256 * 384 * 32) / 256, 256, 0, stream>>>(x, Wih0, b0, Xp);
    // sequential recurrence, persistent: 128 WGs
    gru_seq_kernel<<<128, 256, 0, stream>>>(packW, Xp, bn0, hb, hf, bar, T_SEQ);
    // final h0 lives in hf[0..H) (T even => buffer 0). Layer 1:
    matvec_rows_kernel<<<(H3 * 32) / 256, 256, 0, stream>>>(Wih1, hf, a1, H3, HIDDEN, 1);
    gru_gate0_kernel<<<HIDDEN / 256, 256, 0, stream>>>(a1, b1, bn1, h1);
    // Layer 2:
    matvec_rows_kernel<<<(H3 * 32) / 256, 256, 0, stream>>>(Wih2, h1, a2, H3, HIDDEN, 1);
    gru_gate0_kernel<<<HIDDEN / 256, 256, 0, stream>>>(a2, b2, bn2, h2);
    // Output head:
    out_mv_kernel<<<1, 96, 0, stream>>>(Wout, h2, bout, out);
}
